// Decoder_51866025067223
// MI455X (gfx1250) — matrix operand order repaired and checked by reading, not compiled, not run on hardware
//
#include <hip/hip_runtime.h>
#include <hip/hip_bf16.h>

typedef __attribute__((ext_vector_type(16))) __bf16 bf16x16;
typedef __attribute__((ext_vector_type(8)))  __bf16 bf16x8;
typedef __attribute__((ext_vector_type(8)))  float  floatx8;

#define VOCAB1   32001      // V+1
#define VPAD     32016      // padded to multiple of 16
#define HDIM     1024
#define EDIM     512
#define BATCH    32
#define NSTEP    32
#define G4       4096       // 4*H
#define EOS_TOK  32000

// ---- load two 8x-bf16 chunks into one v16bf operand -------------------------
static __device__ __forceinline__ bf16x16 pack16(const __bf16* p0, const __bf16* p1) {
    bf16x8 lo = *(const bf16x8*)p0;
    bf16x8 hi = *(const bf16x8*)p1;
    bf16x16 r;
#pragma unroll
    for (int i = 0; i < 8; ++i) { r[i] = lo[i]; r[i + 8] = hi[i]; }
    return r;
}

// ---- one K-phase of the tiled GEMM: C[32 x NG*16-ish] += X * W^T ------------
// X: [32, Kx] bf16 row-major.  W: [rows, KW] bf16 row-major; this wave's lane
// uses W rows (rowW0 + g*gate_stride), K window [kw_off, kw_off+Kx).
template <int NG>
static __device__ __forceinline__ void mma_phase(
    const __bf16* __restrict__ X, int Kx,
    const __bf16* __restrict__ W, int KW, int kw_off,
    int rowW0, int gate_stride, int lmod, int lhalf,
    floatx8 (&acc)[2][NG])
{
    for (int kb = 0; kb < Kx; kb += 32) {
        bf16x16 a[2];
#pragma unroll
        for (int t = 0; t < 2; ++t) {
            const __bf16* pa = X + (size_t)(t * 16 + lmod) * Kx + kb + lhalf * 8;
            a[t] = pack16(pa, pa + 16);   // K = base+{0..7}, base+{16..23}
        }
#pragma unroll
        for (int g = 0; g < NG; ++g) {
            const __bf16* pb = W + (size_t)(rowW0 + g * gate_stride) * KW
                                 + kw_off + kb + lhalf * 8;
            bf16x16 b = pack16(pb, pb + 16);  // K = base+{0..7}, base+{16..23}
#pragma unroll
            for (int t = 0; t < 2; ++t)
                acc[t][g] = __builtin_amdgcn_wmma_f32_16x16x32_bf16(
                    false, a[t], false, b, (short)0, acc[t][g], false, false);
        }
    }
}

// ---- fused LSTM layer: gates GEMM + bias + cell update, all in registers ----
// grid 16 blocks x 128 thr (4 waves); each wave owns hidden cols [nbase,nbase+16)
// across all 4 gates and both 16-row M tiles.
__global__ __launch_bounds__(128) void lstm_layer_kernel(
    const __bf16* __restrict__ Xa, int Ka,
    const __bf16* __restrict__ Xb, int Kb,
    const __bf16* __restrict__ W,      // [4096, Ka+Kb] = [w_ih | w_hh]
    const float*  __restrict__ bsum,   // [4096] b_ih + b_hh
    float*        __restrict__ c_state,// [32, 1024] fp32
    __bf16*       __restrict__ h_out)  // [32, 1024] bf16 (next-step input)
{
    const int wave = threadIdx.x >> 5, lane = threadIdx.x & 31;
    const int lmod = lane & 15, lhalf = lane >> 4;
    const int nbase = (blockIdx.x * 4 + wave) * 16;
    const int KW = Ka + Kb;

    floatx8 acc[2][4];
#pragma unroll
    for (int t = 0; t < 2; ++t)
#pragma unroll
        for (int g = 0; g < 4; ++g) acc[t][g] = {};

    mma_phase<4>(Xa, Ka, W, KW, 0,  nbase + lmod, HDIM, lmod, lhalf, acc);
    mma_phase<4>(Xb, Kb, W, KW, Ka, nbase + lmod, HDIM, lmod, lhalf, acc);

    const int hidx = nbase + lmod;
    const float bi = bsum[hidx];
    const float bf = bsum[HDIM + hidx];
    const float bg = bsum[2 * HDIM + hidx];
    const float bo = bsum[3 * HDIM + hidx];

#pragma unroll
    for (int t = 0; t < 2; ++t) {
#pragma unroll
        for (int j = 0; j < 8; ++j) {
            const int m = t * 16 + lhalf * 8 + j;   // C layout: M = j + 8*(lane/16)
            float gi = acc[t][0][j] + bi;
            float gf = acc[t][1][j] + bf;
            float gg = acc[t][2][j] + bg;
            float go = acc[t][3][j] + bo;
            float si = 1.0f / (1.0f + __expf(-gi));
            float sf = 1.0f / (1.0f + __expf(-gf));
            float so = 1.0f / (1.0f + __expf(-go));
            float tg = tanhf(gg);
            float cold = c_state[m * HDIM + hidx];
            float c2 = sf * cold + si * tg;
            c_state[m * HDIM + hidx] = c2;
            h_out[m * HDIM + hidx] = (__bf16)(so * tanhf(c2));
        }
    }
}

// ---- logits projection: [32,1024] x [VPAD,1024]^T + b_out -> d_out ----------
__global__ __launch_bounds__(256) void logits_kernel(
    const __bf16* __restrict__ Xh,     // [32, 1024] bf16 (h of layer 1)
    const __bf16* __restrict__ Wout,   // [VPAD, 1024] bf16 (zero padded)
    const float*  __restrict__ b_out,  // [32001]
    float*        __restrict__ out_step) // [32, 32001]
{
    const int wave = threadIdx.x >> 5, lane = threadIdx.x & 31;
    const int lmod = lane & 15, lhalf = lane >> 4;
    const int nbase = (blockIdx.x * 8 + wave) * 16;
    if (nbase >= VPAD) return;                 // wave-uniform: EXEC stays all-1

    floatx8 acc[2][1];
    acc[0][0] = {}; acc[1][0] = {};
    mma_phase<1>(Xh, HDIM, Wout, HDIM, 0, nbase + lmod, 0, lmod, lhalf, acc);

    const int n = nbase + lmod;
    if (n < VOCAB1) {
        const float bb = b_out[n];
#pragma unroll
        for (int t = 0; t < 2; ++t)
#pragma unroll
            for (int j = 0; j < 8; ++j) {
                const int m = t * 16 + lhalf * 8 + j;
                out_step[(size_t)m * VOCAB1 + n] = acc[t][0][j] + bb;
            }
    }
}

// ---- per-batch argmax (first-max tie-break), EOS/len update, next-token embed
__global__ __launch_bounds__(256) void argmax_finalize_kernel(
    const float* __restrict__ logits_step,   // [32, 32001]
    const float* __restrict__ emb,           // [32001, 512]
    __bf16*      __restrict__ XE,            // [32, 512] next-step embed (bf16)
    int* __restrict__ running, int* __restrict__ len,
    float* __restrict__ len_out, int s, int last)
{
    __shared__ float sv[256];
    __shared__ int   si[256];
    __shared__ int   stok;
    const int b = blockIdx.x, t = threadIdx.x;

    float best = -__builtin_inff();
    int bidx = 0x7fffffff;
    for (int n = t; n < VOCAB1; n += 256) {      // ascending -> keeps first max
        float v = logits_step[(size_t)b * VOCAB1 + n];
        if (v > best) { best = v; bidx = n; }
    }
    sv[t] = best; si[t] = bidx;
    __syncthreads();
    for (int off = 128; off; off >>= 1) {
        if (t < off) {
            float v2 = sv[t + off]; int i2 = si[t + off];
            if (v2 > sv[t] || (v2 == sv[t] && i2 < si[t])) { sv[t] = v2; si[t] = i2; }
        }
        __syncthreads();
    }
    if (t == 0) {
        int tok2 = si[0];
        stok = tok2;
        int r = running[b];
        r = (r && (tok2 != EOS_TOK)) ? 1 : 0;
        running[b] = r;
        int l = len[b];
        if (r) l = s + 1;
        len[b] = l;
        if (last) len_out[b] = (float)l;
    }
    __syncthreads();
    const int tok2 = stok;
    for (int c = t; c < EDIM; c += 256)
        XE[b * EDIM + c] = (__bf16)emb[(size_t)tok2 * EDIM + c];
}

// ---- prep: fp32 -> bf16 weight repack --------------------------------------
__global__ void cvt_cat_kernel(__bf16* __restrict__ dst,
                               const float* __restrict__ a, int wa,
                               const float* __restrict__ b, int wb, long total)
{
    const int wt = wa + wb;
    for (long idx = (long)blockIdx.x * blockDim.x + threadIdx.x; idx < total;
         idx += (long)gridDim.x * blockDim.x) {
        long r = idx / wt; int c = (int)(idx % wt);
        float v = (c < wa) ? a[r * wa + c] : b[r * wb + (c - wa)];
        dst[idx] = (__bf16)v;
    }
}

__global__ void cvt_pad_kernel(__bf16* __restrict__ dst,
                               const float* __restrict__ src,
                               int rows_valid, int width, long total)
{
    for (long idx = (long)blockIdx.x * blockDim.x + threadIdx.x; idx < total;
         idx += (long)gridDim.x * blockDim.x) {
        long r = idx / width;
        dst[idx] = (r < rows_valid) ? (__bf16)src[idx] : (__bf16)0.0f;
    }
}

// ---- prep: biases, states, initial token / hidden buffers -------------------
__global__ __launch_bounds__(256) void init_kernel(
    const float* __restrict__ bi0, const float* __restrict__ bh0,
    const float* __restrict__ bi1, const float* __restrict__ bh1,
    const float* __restrict__ h0, const float* __restrict__ c0,
    const float* __restrict__ emb,
    float* __restrict__ bsum0, float* __restrict__ bsum1,
    float* __restrict__ c_state, __bf16* __restrict__ XE,
    __bf16* __restrict__ XH0a, __bf16* __restrict__ XH1a,
    int* __restrict__ running, int* __restrict__ len)
{
    const int idx = blockIdx.x * 256 + threadIdx.x;    // 65536 threads
    if (idx < G4) { bsum0[idx] = bi0[idx] + bh0[idx]; bsum1[idx] = bi1[idx] + bh1[idx]; }
    if (idx < 2 * BATCH * HDIM) c_state[idx] = c0[idx];
    if (idx < BATCH * EDIM) {
        int c = idx % EDIM;                            // token = SOS for all rows
        XE[idx] = (__bf16)emb[(size_t)EOS_TOK * EDIM + c];
    }
    if (idx < BATCH * HDIM) {
        XH0a[idx] = (__bf16)h0[idx];
        XH1a[idx] = (__bf16)h0[BATCH * HDIM + idx];
    }
    if (idx < BATCH) { running[idx] = 1; len[idx] = 0; }
}

extern "C" void kernel_launch(void* const* d_in, const int* in_sizes, int n_in,
                              void* d_out, int out_size, void* d_ws, size_t ws_size,
                              hipStream_t stream) {
    (void)in_sizes; (void)n_in; (void)out_size; (void)ws_size;
    const float* h0    = (const float*)d_in[2];
    const float* c0    = (const float*)d_in[3];
    const float* emb   = (const float*)d_in[4];
    const float* w_ih0 = (const float*)d_in[5];
    const float* w_hh0 = (const float*)d_in[6];
    const float* b_ih0 = (const float*)d_in[7];
    const float* b_hh0 = (const float*)d_in[8];
    const float* w_ih1 = (const float*)d_in[9];
    const float* w_hh1 = (const float*)d_in[10];
    const float* b_ih1 = (const float*)d_in[11];
    const float* b_hh1 = (const float*)d_in[12];
    const float* w_out = (const float*)d_in[13];
    const float* b_out = (const float*)d_in[14];
    float* out = (float*)d_out;

    // workspace layout (bytes, all 128B-aligned)
    char* ws = (char*)d_ws;
    __bf16* Wcat0 = (__bf16*)(ws + 0);                      // [4096,1536]
    __bf16* Wcat1 = (__bf16*)(ws + 12582912);               // [4096,2048]
    __bf16* Wout  = (__bf16*)(ws + 29360128);               // [32016,1024]
    float*  bsum0 = (float*)(ws + 94928896);                // [4096]
    float*  bsum1 = (float*)(ws + 94945280);                // [4096]
    float*  cst   = (float*)(ws + 94961664);                // [2,32,1024]
    __bf16* XE    = (__bf16*)(ws + 95223808);               // [32,512]
    __bf16* XH0[2] = { (__bf16*)(ws + 95256576), (__bf16*)(ws + 95322112) };
    __bf16* XH1[2] = { (__bf16*)(ws + 95387648), (__bf16*)(ws + 95453184) };
    int* running  = (int*)(ws + 95518720);
    int* lenbuf   = (int*)(ws + 95518848);

    // ---- one-time (per launch) weight repack to bf16 + state init ----------
    cvt_cat_kernel<<<2048, 256, 0, stream>>>(Wcat0, w_ih0, EDIM, w_hh0, HDIM,
                                             (long)G4 * (EDIM + HDIM));
    cvt_cat_kernel<<<2048, 256, 0, stream>>>(Wcat1, w_ih1, HDIM, w_hh1, HDIM,
                                             (long)G4 * (2 * HDIM));
    cvt_pad_kernel<<<2048, 256, 0, stream>>>(Wout, w_out, VOCAB1, HDIM,
                                             (long)VPAD * HDIM);
    init_kernel<<<256, 256, 0, stream>>>(b_ih0, b_hh0, b_ih1, b_hh1, h0, c0, emb,
                                         bsum0, bsum1, cst, XE,
                                         XH0[0], XH1[0], running, lenbuf);

    const size_t step_elems = (size_t)BATCH * VOCAB1;
    float* len_out = out + (size_t)NSTEP * step_elems;

    for (int s = 0; s < NSTEP; ++s) {
        const int r = s & 1, w = r ^ 1;
        lstm_layer_kernel<<<16, 128, 0, stream>>>(XE, EDIM, XH0[r], HDIM,
                                                  Wcat0, bsum0, cst, XH0[w]);
        lstm_layer_kernel<<<16, 128, 0, stream>>>(XH0[w], HDIM, XH1[r], HDIM,
                                                  Wcat1, bsum1, cst + BATCH * HDIM,
                                                  XH1[w]);
        logits_kernel<<<251, 256, 0, stream>>>(XH1[w], Wout, b_out,
                                               out + (size_t)s * step_elems);
        argmax_finalize_kernel<<<BATCH, 256, 0, stream>>>(
            out + (size_t)s * step_elems, emb, XE, running, lenbuf,
            len_out, s, s == NSTEP - 1 ? 1 : 0);
    }
}
